// SimCLRLoss_31507880083995
// MI455X (gfx1250) — matrix operand order repaired and checked by reading, not compiled, not run on hardware
//
#include <hip/hip_runtime.h>

typedef __attribute__((ext_vector_type(16))) __bf16 v16bf;
typedef __attribute__((ext_vector_type(8)))  __bf16 v8bf;
typedef __attribute__((ext_vector_type(8)))  float  v8f;
typedef __attribute__((ext_vector_type(4)))  unsigned int u32x4;
typedef __attribute__((ext_vector_type(8)))  int i32x8;
typedef __attribute__((ext_vector_type(4)))  int i32x4;

#define TWO_N 8192
#define NN    4096
#define DD    512

// ---------------------------------------------------------------------------
// TDM: issue one tensor_load_to_lds moving a 128-row x 32-bf16 tile
// (row stride DD elements) from global into a compact LDS panel.
// D# layout per CDNA5 ISA ch.8 (group0 128b, group1 256b; groups 2/3 unused).
// ---------------------------------------------------------------------------
__device__ __forceinline__ void tdm_load_tile_128x32(const __bf16* gsrc,
                                                     __bf16* ldst) {
    const unsigned long long ga = (unsigned long long)(uintptr_t)gsrc;
    const unsigned int la = (unsigned int)(uintptr_t)ldst; // LDS aperture: [31:0] = LDS byte addr

    u32x4 g0;
    g0[0] = 1u;                                   // count=1, user mode, no gather
    g0[1] = la;                                   // lds_addr
    g0[2] = (unsigned int)ga;                     // global_addr[31:0]
    g0[3] = (unsigned int)((ga >> 32) & 0x01FFFFFFu) | (2u << 30); // addr[56:32], type=2

    i32x8 g1;
    g1[0] = (int)(1u << 16);                      // workgroup_mask=0, data_size=1 (2 bytes)
    g1[1] = (int)((unsigned)(DD & 0xFFFF) << 16); // tensor_dim0[15:0]=512 -> bits 63:48
    g1[2] = (int)(((unsigned)DD >> 16) | ((unsigned)(TWO_N & 0xFFFF) << 16)); // dim0 hi | dim1 lo
    g1[3] = (int)(((unsigned)TWO_N >> 16) | (32u << 16)); // dim1 hi | tile_dim0=32
    g1[4] = (int)128u;                            // tile_dim1=128, tile_dim2=0
    g1[5] = (int)DD;                              // tensor_dim0_stride=512 (low 32)
    g1[6] = 0;                                    // stride hi | tensor_dim1_stride lo
    g1[7] = 0;

    const i32x4 gz = {0, 0, 0, 0};
#if defined(__clang_major__) && (__clang_major__ >= 23)
    const i32x8 gz8 = {0, 0, 0, 0, 0, 0, 0, 0};
    __builtin_amdgcn_tensor_load_to_lds(g0, g1, gz, gz, gz8, 0);
#else
    __builtin_amdgcn_tensor_load_to_lds(g0, g1, gz, gz, 0);
#endif
}

// ---------------------------------------------------------------------------
// Kernel 1: row-normalize to unit L2 norm, split into bf16 hi/lo pair.
// One wave (32 lanes) per row. Also zeroes rowsum (first 32 blocks).
// ---------------------------------------------------------------------------
__global__ void simclr_normalize_kernel(const float* __restrict__ X,
                                        __bf16* __restrict__ Xhi,
                                        __bf16* __restrict__ Xlo,
                                        float* __restrict__ rowsum) {
    const int wave = threadIdx.x >> 5;
    const int lane = threadIdx.x & 31;
    const int row  = blockIdx.x * 8 + wave;

    if (blockIdx.x < 32) rowsum[blockIdx.x * 256 + threadIdx.x] = 0.0f;

    const float* xr = X + (size_t)row * DD;
    float v[16];
    float ss = 0.0f;
#pragma unroll
    for (int i = 0; i < 16; ++i) {
        v[i] = xr[lane + i * 32];
        ss += v[i] * v[i];
    }
#pragma unroll
    for (int off = 16; off >= 1; off >>= 1) ss += __shfl_xor(ss, off, 32);
    const float inv = rsqrtf(ss);

    __bf16* hr = Xhi + (size_t)row * DD;
    __bf16* lr = Xlo + (size_t)row * DD;
#pragma unroll
    for (int i = 0; i < 16; ++i) {
        const float xn = v[i] * inv;
        const __bf16 h = (__bf16)xn;
        const float  lo = xn - (float)h;
        hr[lane + i * 32] = h;
        lr[lane + i * 32] = (__bf16)lo;
    }
}

// ---------------------------------------------------------------------------
// Kernel 2: Gram matrix via bf16 WMMA (hi/lo compensated ~fp32 accuracy),
// TDM double-buffered LDS staging, fused exp() + row-sum epilogue.
// Each block computes a 128x128 tile of S = Xn*Xn^T; 8 waves, each wave
// owns a 16x128 strip (8 f32 16x16 accumulators).
// ---------------------------------------------------------------------------
__global__ void __launch_bounds__(256)
simclr_gram_kernel(const __bf16* __restrict__ Xhi,
                   const __bf16* __restrict__ Xlo,
                   float* __restrict__ rowsum) {
    __shared__ __attribute__((aligned(32))) __bf16 sAh[2][128 * 32];
    __shared__ __attribute__((aligned(32))) __bf16 sAl[2][128 * 32];
    __shared__ __attribute__((aligned(32))) __bf16 sBh[2][128 * 32];
    __shared__ __attribute__((aligned(32))) __bf16 sBl[2][128 * 32];

    const int rowBase = blockIdx.y * 128;
    const int colBase = blockIdx.x * 128;
    const int wave = threadIdx.x >> 5;
    const int lane = threadIdx.x & 31;

    const __bf16* gAh = Xhi + (size_t)rowBase * DD;
    const __bf16* gAl = Xlo + (size_t)rowBase * DD;
    const __bf16* gBh = Xhi + (size_t)colBase * DD;
    const __bf16* gBl = Xlo + (size_t)colBase * DD;

    v8f acc[8];
#pragma unroll
    for (int i = 0; i < 8; ++i) acc[i] = (v8f){0.f, 0.f, 0.f, 0.f, 0.f, 0.f, 0.f, 0.f};

    // fragment addressing (contiguous K order, used alike by A and B)
    const int m  = lane & 15;        // A row / B col within 16-wide tile
    const int kh = (lane >> 4) * 8;  // unused: A uses the same contiguous K order as B (kb)
    const int kb = (lane >> 4) * 16; // B: lanes 0-15 hold K 0..15, lanes 16-31 K 16..31

    // Prologue: TDM-stage K-slice 0 into buffer 0 (wave 0 issues; TENSORcnt tracks).
    if (wave == 0) {
        tdm_load_tile_128x32(gAh, &sAh[0][0]);
        tdm_load_tile_128x32(gAl, &sAl[0][0]);
        tdm_load_tile_128x32(gBh, &sBh[0][0]);
        tdm_load_tile_128x32(gBl, &sBl[0][0]);
    }

    for (int it = 0; it < DD / 32; ++it) {
        const int cur = it & 1;
        if (wave == 0) {
            if (it + 1 < DD / 32) {
                const int nxt = cur ^ 1;
                const int k0  = (it + 1) * 32;
                tdm_load_tile_128x32(gAh + k0, &sAh[nxt][0]);
                tdm_load_tile_128x32(gAl + k0, &sAl[nxt][0]);
                tdm_load_tile_128x32(gBh + k0, &sBh[nxt][0]);
                tdm_load_tile_128x32(gBl + k0, &sBl[nxt][0]);
                // TDM completes in order: <=4 outstanding => current buffer done.
                __builtin_amdgcn_s_wait_tensorcnt(4);
            } else {
                __builtin_amdgcn_s_wait_tensorcnt(0);
            }
        }
        __syncthreads();  // publish TDM-written LDS panel to all waves

        // A fragments (this wave's 16-row strip)
        const int abase = (wave * 16 + m) * 32;
        v16bf Ah = *(const v16bf*)&sAh[cur][abase + kb];
        v16bf Al = *(const v16bf*)&sAl[cur][abase + kb];

#pragma unroll
        for (int tt = 0; tt < 8; ++tt) {
            // B column n of the S-tile == row (colBase + tt*16 + n) of Xn: no transpose.
            v16bf Bh = *(const v16bf*)&sBh[cur][(tt * 16 + m) * 32 + kb];
            v16bf Bl = *(const v16bf*)&sBl[cur][(tt * 16 + m) * 32 + kb];
            acc[tt] = __builtin_amdgcn_wmma_f32_16x16x32_bf16(
                false, Ah, false, Bh, (short)0, acc[tt], false, false);
            acc[tt] = __builtin_amdgcn_wmma_f32_16x16x32_bf16(
                false, Ah, false, Bl, (short)0, acc[tt], false, false);
            acc[tt] = __builtin_amdgcn_wmma_f32_16x16x32_bf16(
                false, Al, false, Bh, (short)0, acc[tt], false, false);
        }
        __syncthreads();  // all reads of buf[cur] done before TDM rewrites it
    }

    // Fused epilogue: rowsum[r] += sum_cols exp(sim).
    // C/D layout: lane L holds col N=L&15; VGPR j holds row M = j + 8*(L>>4).
#pragma unroll
    for (int j = 0; j < 8; ++j) {
        float s = 0.0f;
#pragma unroll
        for (int tt = 0; tt < 8; ++tt) s += __expf(acc[tt][j]);
        s += __shfl_xor(s, 1, 32);
        s += __shfl_xor(s, 2, 32);
        s += __shfl_xor(s, 4, 32);
        s += __shfl_xor(s, 8, 32);
        const int rowM = rowBase + wave * 16 + j + 8 * (lane >> 4);
        if ((lane & 15) == 0) unsafeAtomicAdd(&rowsum[rowM], s);
    }
}

// ---------------------------------------------------------------------------
// Kernel 3: per-row fp32 pair-dot (log numerator) and self-dot (diagonal).
// One wave per row.
// ---------------------------------------------------------------------------
__global__ void simclr_pair_kernel(const __bf16* __restrict__ Xhi,
                                   const __bf16* __restrict__ Xlo,
                                   float* __restrict__ pairdot,
                                   float* __restrict__ selfdot) {
    const int wave = threadIdx.x >> 5;
    const int lane = threadIdx.x & 31;
    const int row  = blockIdx.x * 8 + wave;
    const int prow = (row < NN) ? row + NN : row - NN;

    const __bf16* hr = Xhi + (size_t)row  * DD;
    const __bf16* lr = Xlo + (size_t)row  * DD;
    const __bf16* hp = Xhi + (size_t)prow * DD;
    const __bf16* lp = Xlo + (size_t)prow * DD;

    float pd = 0.0f, sd = 0.0f;
#pragma unroll
    for (int i = lane; i < DD; i += 32) {
        const float xi = (float)hr[i] + (float)lr[i];
        const float xp = (float)hp[i] + (float)lp[i];
        pd += xi * xp;
        sd += xi * xi;
    }
#pragma unroll
    for (int off = 16; off >= 1; off >>= 1) {
        pd += __shfl_xor(pd, off, 32);
        sd += __shfl_xor(sd, off, 32);
    }
    if (lane == 0) {
        pairdot[row] = pd;
        selfdot[row] = sd;
    }
}

// ---------------------------------------------------------------------------
// Kernel 4: loss = mean_i [ log(rowsum_i - exp(selfdot_i)) - pairdot_i ]
// ---------------------------------------------------------------------------
__global__ void simclr_finalize_kernel(const float* __restrict__ rowsum,
                                       const float* __restrict__ pairdot,
                                       const float* __restrict__ selfdot,
                                       float* __restrict__ out) {
    __shared__ float red[256];
    float s = 0.0f;
    for (int i = threadIdx.x; i < TWO_N; i += 256) {
        s += logf(rowsum[i] - __expf(selfdot[i])) - pairdot[i];
    }
    red[threadIdx.x] = s;
    __syncthreads();
    for (int off = 128; off >= 1; off >>= 1) {
        if (threadIdx.x < off) red[threadIdx.x] += red[threadIdx.x + off];
        __syncthreads();
    }
    if (threadIdx.x == 0) out[0] = red[0] / (float)TWO_N;
}

// ---------------------------------------------------------------------------
extern "C" void kernel_launch(void* const* d_in, const int* in_sizes, int n_in,
                              void* d_out, int out_size, void* d_ws, size_t ws_size,
                              hipStream_t stream) {
    (void)in_sizes; (void)n_in; (void)out_size; (void)ws_size;
    const float* X = (const float*)d_in[0];

    char* ws = (char*)d_ws;
    __bf16* Xhi    = (__bf16*)ws;                                   // 8 MB
    __bf16* Xlo    = (__bf16*)(ws + (size_t)TWO_N * DD * 2);        // 8 MB
    float*  rowsum = (float*)(ws + (size_t)TWO_N * DD * 4);         // 32 KB
    float*  pairdot = rowsum + TWO_N;
    float*  selfdot = pairdot + TWO_N;

    simclr_normalize_kernel<<<TWO_N / 8, 256, 0, stream>>>(X, Xhi, Xlo, rowsum);

    dim3 grid(TWO_N / 128, TWO_N / 128);
    simclr_gram_kernel<<<grid, 256, 0, stream>>>(Xhi, Xlo, rowsum);

    simclr_pair_kernel<<<TWO_N / 8, 256, 0, stream>>>(Xhi, Xlo, pairdot, selfdot);

    simclr_finalize_kernel<<<1, 256, 0, stream>>>(rowsum, pairdot, selfdot,
                                                  (float*)d_out);
}
